// Conv2d_lr_56899726737745
// MI455X (gfx1250) — matrix operand order repaired and checked by reading, not compiled, not run on hardware
//
#include <hip/hip_runtime.h>

// ---------------------------------------------------------------------------
// Low-rank 3x3 conv on gfx1250 (MI455X), wave32 + V_WMMA_F32_16X16X32_F16,
// with TDM (tensor_load_to_lds) double-buffered staging of the folded weight
// W = V S^T into LDS.  Halo handled branchlessly via a zero pad page.
//   out = unfold(x) . (V S^T) . U^T + bias
// Sizes: B=16, C=256, H=W=56, L=3136, r=64, O=256, Ktot=2304.
// ---------------------------------------------------------------------------

typedef __attribute__((ext_vector_type(16))) _Float16     v16h;
typedef __attribute__((ext_vector_type(8)))  float        v8f;
typedef __attribute__((ext_vector_type(4)))  unsigned int v4u;
typedef __attribute__((ext_vector_type(4)))  int          v4i;
typedef __attribute__((ext_vector_type(8)))  int          v8i;

#define NB   16
#define NC   256
#define NHW  3136         // 56*56
#define NO   256
#define MTOT (NB * NHW)   // 50176 = 256 * 196

// workspace offsets (bytes)
#define OFF_XH   0u                 // f16 [B, L, C]          25,690,112 B
#define OFF_WP   25690112u          // f16 fragments, 294,912 B
#define OFF_UP   25985024u          // f16 fragments,  32,768 B
#define OFF_Z    26017792u          // f16 [MTOT, 64]  6,422,528 B
#define OFF_ZPAD 32440320u          // 512 B of zeros (halo reads)

union AFrag { v16h v; uint4 q[2]; };

#if defined(__gfx1250__) && __has_builtin(__builtin_amdgcn_tensor_load_to_lds) && \
    __has_builtin(__builtin_amdgcn_s_wait_tensorcnt)
#define USE_TDM 1
#endif

// ---------------------------------------------------------------------------
// TDM: async DMA of one contiguous 32KB slab (global -> LDS).
// D# per CDNA5 ISA:
//   group0: count=1 | lds_addr | global_addr | type=2
//   group1: data_size=3 (8B units), tensor_dim0=tile_dim0=4096, dim1=1
// ---------------------------------------------------------------------------
__device__ __forceinline__ void tdm_load_slab32k(const void* gsrc, void* lds_dst) {
#ifdef USE_TDM
    unsigned int lds_off = (unsigned int)(size_t)lds_dst;  // low 32b of flat = LDS offset
    unsigned long long ga = (unsigned long long)(size_t)gsrc;
    v4u g0 = { 1u,
               lds_off,
               (unsigned int)ga,
               (unsigned int)((ga >> 32) & 0x01FFFFFFu) | (2u << 30) };
    v8i g1 = { (int)0x00030000u,                      // wg_mask=0, data_size=3 (8B)
               (int)0x10000000u,                      // tensor_dim0[15:0]=4096 in [31:16]
               (int)0x00010000u,                      // tensor_dim0 hi=0, tensor_dim1=1
               (int)0x10000000u,                      // tensor_dim1 hi=0, tile_dim0=4096
               (int)0x00000001u,                      // tile_dim1=1, tile_dim2=0
               (int)0x00001000u,                      // tensor_dim0_stride=4096
               0, 0 };
    v4i g2 = { 0, 0, 0, 0 };
    v4i g3 = { 0, 0, 0, 0 };
#if __clang_major__ >= 23
    v8i g4 = { 0, 0, 0, 0, 0, 0, 0, 0 };
    __builtin_amdgcn_tensor_load_to_lds(g0, g1, g2, g3, g4, 0);
#else
    __builtin_amdgcn_tensor_load_to_lds(g0, g1, g2, g3, 0);
#endif
#else
    (void)gsrc; (void)lds_dst;
#endif
}

// ---------------------------------------------------------------------------
// Kernel 1: build W = V @ S^T (k2 = pos*256 + chan ordering), pack W and U
// into per-lane WMMA B-fragment layout (f16 32x16), and zero the halo pad.
// ---------------------------------------------------------------------------
__global__ __launch_bounds__(256) void pack_kernel(const float* __restrict__ U,
                                                   const float* __restrict__ S,
                                                   const float* __restrict__ V,
                                                   _Float16* __restrict__ Wp,
                                                   _Float16* __restrict__ Up,
                                                   _Float16* __restrict__ zpad) {
    int id = blockIdx.x * 256 + threadIdx.x;
    if (id < 72 * 4 * 512) {                       // W fragments
        int e    = id & 15;
        int lane = (id >> 4) & 31;
        int frag = id >> 9;
        int nb   = frag & 3;
        int kc   = frag >> 2;                      // 0..71
        int kB   = (e & 7) + ((lane >> 4) << 3) + ((e >> 3) << 4);
        int k2   = kc * 32 + kB;
        int rr   = nb * 16 + (lane & 15);
        int p    = k2 >> 8;
        int c    = k2 & 255;
        int k1   = c * 9 + p;                      // torch unfold row ordering
        float acc = 0.f;
        for (int rp = 0; rp < 64; ++rp)
            acc += V[k1 * 64 + rp] * S[rr * 64 + rp];   // W = V @ S^T
        Wp[id] = (_Float16)acc;
    } else {
        int id2 = id - 72 * 4 * 512;
        if (id2 < 2 * 16 * 512) {                  // U fragments
            int e    = id2 & 15;
            int lane = (id2 >> 4) & 31;
            int frag = id2 >> 9;
            int nb   = frag & 15;
            int kc   = frag >> 4;
            int k    = kc * 32 + (e & 7) + ((lane >> 4) << 3) + ((e >> 3) << 4);
            int o    = nb * 16 + (lane & 15);
            Up[id2] = (_Float16)U[o * 64 + k];
        } else {
            int id3 = id2 - 2 * 16 * 512;
            if (id3 < 256) zpad[id3] = (_Float16)0.f;   // 512 B zero page
        }
    }
}

// ---------------------------------------------------------------------------
// Kernel 2: x NCHW f32 -> xh NHWC f16 via LDS 32x32 tile transpose.
// ---------------------------------------------------------------------------
__global__ __launch_bounds__(256) void convert_kernel(const float* __restrict__ x,
                                                      _Float16* __restrict__ xh) {
    __shared__ float tile[32][33];
    const int tx = threadIdx.x, ty = threadIdx.y;
    const int l0 = blockIdx.x * 32;
    const int c0 = blockIdx.y * 32;
    const int b  = blockIdx.z;
#pragma unroll
    for (int k = 0; k < 4; ++k) {
        int c = c0 + ty + k * 8;
        tile[ty + k * 8][tx] = x[((size_t)(b * NC + c)) * NHW + l0 + tx];
    }
    __syncthreads();
#pragma unroll
    for (int k = 0; k < 4; ++k) {
        int lp = l0 + ty + k * 8;
        xh[((size_t)(b * NHW + lp)) * NC + c0 + tx] = (_Float16)tile[tx][ty + k * 8];
    }
}

// ---------------------------------------------------------------------------
// Kernel 3: stage-1 implicit conv GEMM.  z[m, 0..63] = sum_k cols[m,k]*W[k,:]
// grid 196, block 256 (8 waves).  Wave owns 32 M-rows x 64 N.  W slab for
// kernel position p (32KB) staged into LDS by TDM, double buffered.
// Halo rows read the zero page -> no exec-mask branches in the hot loop.
// ---------------------------------------------------------------------------
__global__ __launch_bounds__(256) void stage1_kernel(const _Float16* __restrict__ xh,
                                                     const _Float16* __restrict__ Wp,
                                                     const _Float16* __restrict__ zpad,
                                                     _Float16* __restrict__ z) {
    __shared__ __align__(64) char smem[65536];     // 2x32KB W slabs; reused for z
    const int tid  = threadIdx.x;
    const int lane = tid & 31;
    const int wave = tid >> 5;
    const int m0   = blockIdx.x * 256 + wave * 32;
    const int koff = (lane >> 4) * 8;
    const int ncol = lane & 15;

    const int mA   = m0 + ncol;                    // rows 0..15 of wave tile
    const int mB   = mA + 16;                      // rows 16..31 (same batch: 32|3136)
    const int b    = mA / NHW;
    const int pixA = mA - b * NHW;
    const int hA   = pixA / 56, wA = pixA - hA * 56;
    const int pixB = mB - b * NHW;
    const int hB   = pixB / 56, wB = pixB - hB * 56;

    v8f ac0 = {}, ac1 = {}, ac2 = {}, ac3 = {};    // rows 0..15
    v8f bc0 = {}, bc1 = {}, bc2 = {}, bc3 = {};    // rows 16..31

    // ---- stage W slab for p=0 ----
#ifdef USE_TDM
    if (wave == 0) tdm_load_slab32k(Wp, smem);
#else
    {
        const uint4* s = (const uint4*)Wp;
        uint4* d = (uint4*)smem;
#pragma unroll
        for (int i = 0; i < 8; ++i) d[tid + i * 256] = s[tid + i * 256];
    }
#endif

    for (int p = 0; p < 9; ++p) {
        if (p + 1 < 9) {                           // prefetch next slab
#ifdef USE_TDM
            if (wave == 0)
                tdm_load_slab32k(Wp + (size_t)(p + 1) * 16384, smem + ((p + 1) & 1) * 32768);
#else
            const uint4* s = (const uint4*)(Wp + (size_t)(p + 1) * 16384);
            uint4* d = (uint4*)(smem + ((p + 1) & 1) * 32768);
#pragma unroll
            for (int i = 0; i < 8; ++i) d[tid + i * 256] = s[tid + i * 256];
#endif
        }
#ifdef USE_TDM
        if (wave == 0) {
            if (p + 1 < 9) __builtin_amdgcn_s_wait_tensorcnt(1);  // slab p done
            else           __builtin_amdgcn_s_wait_tensorcnt(0);
        }
#endif
        __syncthreads();                           // slab p visible to all waves

        const int dh = (p / 3) - 1, dw = (p % 3) - 1;
        const int hhA = hA + dh, wwA = wA + dw;
        const int hhB = hB + dh, wwB = wB + dw;
        const bool vA = (hhA >= 0) && (hhA < 56) && (wwA >= 0) && (wwA < 56);
        const bool vB = (hhB >= 0) && (hhB < 56) && (wwB >= 0) && (wwB < 56);
        const int idxA = vA ? (b * NHW + hhA * 56 + wwA) : 0;
        const int idxB = vB ? (b * NHW + hhB * 56 + wwB) : 0;
        const _Float16* apA = (vA ? xh + (size_t)idxA * NC : zpad) + koff;
        const _Float16* apB = (vB ? xh + (size_t)idxB * NC : zpad) + koff;
        const _Float16* wbase = (const _Float16*)(smem + (p & 1) * 32768) + lane * 16;

#pragma unroll
        for (int cb = 0; cb < 8; ++cb) {
            AFrag a0, a1;
            a0.q[0] = *(const uint4*)(apA + cb * 32);
            a0.q[1] = *(const uint4*)(apA + cb * 32 + 16);
            a1.q[0] = *(const uint4*)(apB + cb * 32);
            a1.q[1] = *(const uint4*)(apB + cb * 32 + 16);
            const _Float16* wf = wbase + cb * 2048;
            v16h b0 = *(const v16h*)(wf);
            v16h b1 = *(const v16h*)(wf + 512);
            v16h b2 = *(const v16h*)(wf + 1024);
            v16h b3 = *(const v16h*)(wf + 1536);
            ac0 = __builtin_amdgcn_wmma_f32_16x16x32_f16(false, a0.v, false, b0, (short)0, ac0, false, false);
            bc0 = __builtin_amdgcn_wmma_f32_16x16x32_f16(false, a1.v, false, b0, (short)0, bc0, false, false);
            ac1 = __builtin_amdgcn_wmma_f32_16x16x32_f16(false, a0.v, false, b1, (short)0, ac1, false, false);
            bc1 = __builtin_amdgcn_wmma_f32_16x16x32_f16(false, a1.v, false, b1, (short)0, bc1, false, false);
            ac2 = __builtin_amdgcn_wmma_f32_16x16x32_f16(false, a0.v, false, b2, (short)0, ac2, false, false);
            bc2 = __builtin_amdgcn_wmma_f32_16x16x32_f16(false, a1.v, false, b2, (short)0, bc2, false, false);
            ac3 = __builtin_amdgcn_wmma_f32_16x16x32_f16(false, a0.v, false, b3, (short)0, ac3, false, false);
            bc3 = __builtin_amdgcn_wmma_f32_16x16x32_f16(false, a1.v, false, b3, (short)0, bc3, false, false);
        }
        __syncthreads();                           // done reading slab p
    }

    // ---- epilogue: transpose D tiles in LDS (row stride 72 halves = 144B),
    //      then coalesced row-major f16 stores of z ----
    _Float16* lz = (_Float16*)smem;
    const int woff = wave * 2304;                  // 32 rows * 72 halves
    const int rext = (lane >> 4) * 8;
#pragma unroll
    for (int g = 0; g < 8; ++g) {
        const int rA = (g + rext) * 72 + ncol;
        const int rB = (g + rext + 16) * 72 + ncol;
        lz[woff + rA +  0] = (_Float16)ac0[g];
        lz[woff + rA + 16] = (_Float16)ac1[g];
        lz[woff + rA + 32] = (_Float16)ac2[g];
        lz[woff + rA + 48] = (_Float16)ac3[g];
        lz[woff + rB +  0] = (_Float16)bc0[g];
        lz[woff + rB + 16] = (_Float16)bc1[g];
        lz[woff + rB + 32] = (_Float16)bc2[g];
        lz[woff + rB + 48] = (_Float16)bc3[g];
    }
    __syncthreads();

    const uint4* src = (const uint4*)(lz + woff + lane * 72);   // 144B stride, 16B aligned
    uint4* zg = (uint4*)(z + ((size_t)(m0 + lane)) * 64);
#pragma unroll
    for (int q = 0; q < 8; ++q)
        zg[q] = src[q];
}

// ---------------------------------------------------------------------------
// Kernel 4: stage-2 GEMM out[m,o] = z[m,:] . U[o,:] + bias[o], NCHW f32 out.
// grid (392, 4), block 256.  Wave: 16 M-rows x 64 O-cols, K=64.
// ---------------------------------------------------------------------------
__global__ __launch_bounds__(256) void stage2_kernel(const _Float16* __restrict__ z,
                                                     const _Float16* __restrict__ Up,
                                                     const float* __restrict__ bias,
                                                     float* __restrict__ out) {
    __shared__ float lt[8 * 64 * 20];              // 40 KB
    const int lane = threadIdx.x & 31;
    const int wave = threadIdx.x >> 5;
    const int m0   = blockIdx.x * 128 + wave * 16;
    const int o0   = blockIdx.y * 64;
    const int koff = (lane >> 4) * 8;
    const int m    = m0 + (lane & 15);

    v8f acc0 = {}, acc1 = {}, acc2 = {}, acc3 = {};
#pragma unroll
    for (int kc = 0; kc < 2; ++kc) {
        AFrag a;
        const _Float16* zp = z + (size_t)m * 64 + kc * 32 + koff;
        a.q[0] = *(const uint4*)(zp);
        a.q[1] = *(const uint4*)(zp + 16);
        const _Float16* uf = Up + ((size_t)(kc * 16 + blockIdx.y * 4)) * 512 + lane * 16;
        v16h b0 = *(const v16h*)(uf);
        v16h b1 = *(const v16h*)(uf + 512);
        v16h b2 = *(const v16h*)(uf + 1024);
        v16h b3 = *(const v16h*)(uf + 1536);
        acc0 = __builtin_amdgcn_wmma_f32_16x16x32_f16(false, a.v, false, b0, (short)0, acc0, false, false);
        acc1 = __builtin_amdgcn_wmma_f32_16x16x32_f16(false, a.v, false, b1, (short)0, acc1, false, false);
        acc2 = __builtin_amdgcn_wmma_f32_16x16x32_f16(false, a.v, false, b2, (short)0, acc2, false, false);
        acc3 = __builtin_amdgcn_wmma_f32_16x16x32_f16(false, a.v, false, b3, (short)0, acc3, false, false);
    }

    const int ncol = lane & 15;
    acc0 += bias[o0 + ncol];
    acc1 += bias[o0 + 16 + ncol];
    acc2 += bias[o0 + 32 + ncol];
    acc3 += bias[o0 + 48 + ncol];

    // LDS col-major (stride 20 dwords -> 16B aligned, bank-spread)
    const int lbase = wave * 1280;
    const int rext  = (lane >> 4) * 8;
#pragma unroll
    for (int g = 0; g < 8; ++g) {
        const int rr = g + rext;
        lt[lbase + (ncol +  0) * 20 + rr] = acc0[g];
        lt[lbase + (ncol + 16) * 20 + rr] = acc1[g];
        lt[lbase + (ncol + 32) * 20 + rr] = acc2[g];
        lt[lbase + (ncol + 48) * 20 + rr] = acc3[g];
    }
    __syncthreads();

    const int b    = m0 / NHW;
    const int pix0 = m0 - b * NHW;
    float* outp = out + ((size_t)b * NO + o0) * NHW;
#pragma unroll
    for (int t = 0; t < 8; ++t) {
        const int chunk = t * 32 + lane;
        const int col   = chunk >> 2;
        const int q     = chunk & 3;
        uint4 v = *(const uint4*)&lt[lbase + col * 20 + q * 4];
        *(uint4*)&outp[(size_t)col * NHW + pix0 + q * 4] = v;
    }
}

// ---------------------------------------------------------------------------
extern "C" void kernel_launch(void* const* d_in, const int* in_sizes, int n_in,
                              void* d_out, int out_size, void* d_ws, size_t ws_size,
                              hipStream_t stream) {
    const float* x    = (const float*)d_in[0];
    const float* U    = (const float*)d_in[1];
    const float* S    = (const float*)d_in[2];
    const float* V    = (const float*)d_in[3];
    const float* bias = (const float*)d_in[4];

    char* ws = (char*)d_ws;
    _Float16* xh   = (_Float16*)(ws + OFF_XH);
    _Float16* Wp   = (_Float16*)(ws + OFF_WP);
    _Float16* Up   = (_Float16*)(ws + OFF_UP);
    _Float16* zb   = (_Float16*)(ws + OFF_Z);
    _Float16* zpad = (_Float16*)(ws + OFF_ZPAD);

    pack_kernel<<<641, 256, 0, stream>>>(U, S, V, Wp, Up, zpad);
    convert_kernel<<<dim3(98, 8, 16), dim3(32, 8), 0, stream>>>(x, xh);
    stage1_kernel<<<196, 256, 0, stream>>>(xh, Wp, zpad, zb);
    stage2_kernel<<<dim3(392, 4), 256, 0, stream>>>(zb, Up, bias, (float*)d_out);
}
